// InducedSetAttentionMini_17171279250121
// MI455X (gfx1250) — matrix operand order repaired and checked by reading, not compiled, not run on hardware
//
#include <hip/hip_runtime.h>

// ---------------------------------------------------------------------------
// Induced Set Attention (B=8,H=8,Lq=Lk=4096,LI=256,D=64) for gfx1250 (MI455X)
// f16-operand / f32-accumulate WMMA (v_wmma_f32_16x16x32_f16), wave32.
// Round 2: register-cache A/B fragments to cut redundant LDS traffic.
// ---------------------------------------------------------------------------

typedef _Float16 h8   __attribute__((ext_vector_type(8)));
typedef _Float16 v16h __attribute__((ext_vector_type(16)));
typedef float    v8f  __attribute__((ext_vector_type(8)));

#define BH_     64
#define LI_     256
#define LK_     4096
#define LQ_     4096
#define D_      64
#define DH_     512
#define KSPLIT  8
#define KCHUNK  (LK_ / KSPLIT)   // 512
#define KT      64               // k rows staged per iteration
#define PAD     72               // row stride in halves (144B, 16B multiple)
#define PADI    264              // row stride for LI-wide rows (528B, 16B mult)

__device__ __forceinline__ v8f wmma_f16(v16h a, v16h b, v8f c) {
  return __builtin_amdgcn_wmma_f32_16x16x32_f16(false, a, false, b,
                                                (short)0, c, false, false);
}

// A-fragment 16x32 f16: lanes 0-15 hold M=lane, K{0..7,16..23};
// lanes 16-31 hold M=lane-16, K{8..15,24..31}.  rowk = row base (+k offset).
__device__ __forceinline__ v16h frag_a(const _Float16* rowk, int lane) {
  const int klo = (lane & 16) ? 8 : 0;
  h8 lo = *(const h8*)(rowk + klo);
  h8 hi = *(const h8*)(rowk + 16 + klo);
  return __builtin_shufflevector(lo, hi, 0,1,2,3,4,5,6,7,8,9,10,11,12,13,14,15);
}

// B-fragment 32x16 f16: lanes 0-15 hold N=lane, K{0..7,16..23};
// lanes 16-31 hold N=lane-16, K{8..15,24..31}.
// rown = source row (B column) base pointer (+k offset).
__device__ __forceinline__ v16h frag_b(const _Float16* rown, int lane) {
  const int kb = (lane & 16) ? 8 : 0;
  h8 lo = *(const h8*)(rown + kb);
  h8 hi = *(const h8*)(rown + 16 + kb);
  return __builtin_shufflevector(lo, hi, 0,1,2,3,4,5,6,7,8,9,10,11,12,13,14,15);
}

__device__ __forceinline__ float relu15f(float x) {
  return fminf(fmaxf(x, 0.0f), 1e15f);
}

// ---------------------------------------------------------------------------
// Stage 0: Hraw[bh][i][d] += relu15((I/8) @ k^T) @ v   (k-split partials)
// grid (KSPLIT, BH), block 256 (8 waves).  Dyn LDS = 92160 B.
// ---------------------------------------------------------------------------
__global__ void __launch_bounds__(256) isab_stage0(
    const float* __restrict__ I, const float* __restrict__ K,
    const float* __restrict__ V, float* __restrict__ Hraw)
{
  extern __shared__ _Float16 smem[];
  _Float16* sI  = smem;               // [256][PAD]  I/8 (f16)
  _Float16* sK  = sI  + LI_ * PAD;    // [64][PAD]   k tile
  _Float16* sVT = sK  + KT  * PAD;    // [64][PAD]   v tile transposed (d, kk)
  _Float16* sS  = sVT + KT  * PAD;    // [8][32][PAD] per-wave S (f16)

  const int bh   = blockIdx.y;
  const int kc   = blockIdx.x;
  const int tid  = threadIdx.x;
  const int lane = tid & 31;
  const int wave = tid >> 5;

  const float* Ip = I + (size_t)bh * LI_ * D_;
  const float* Kp = K + (size_t)bh * LK_ * D_ + (size_t)kc * KCHUNK * D_;
  const float* Vp = V + (size_t)bh * LK_ * D_ + (size_t)kc * KCHUNK * D_;

  // stage I/8 as f16
  for (int idx = tid; idx < LI_ * D_; idx += 256) {
    int r = idx >> 6, c = idx & 63;
    sI[r * PAD + c] = (_Float16)(Ip[idx] * 0.125f);
  }

  v8f acc[2][4];
  v8f vzero = {0.f,0.f,0.f,0.f,0.f,0.f,0.f,0.f};
  #pragma unroll
  for (int i = 0; i < 2; ++i)
    #pragma unroll
    for (int j = 0; j < 4; ++j) acc[i][j] = vzero;

  const int m0 = wave * 32;                 // wave's 32-row M block of LI
  _Float16* sSw = sS + wave * 32 * PAD;

  for (int kk0 = 0; kk0 < KCHUNK; kk0 += KT) {
    __syncthreads();
    // stage k tile (row-major) and v tile transposed, f32 -> f16
    for (int idx = tid; idx < KT * D_; idx += 256) {
      int r = idx >> 6, c = idx & 63;
      sK[r * PAD + c]  = (_Float16)Kp[(size_t)(kk0 + r) * D_ + c];
      sVT[c * PAD + r] = (_Float16)Vp[(size_t)(kk0 + r) * D_ + c];
    }
    // prefetch next tile (speculative; dropped if OOB)
    __builtin_prefetch(Kp + (size_t)(kk0 + KT) * D_ + tid, 0, 1);
    __builtin_prefetch(Vp + (size_t)(kk0 + KT) * D_ + tid, 0, 1);
    __syncthreads();

    // --- S = (I/8) @ k^T : rows [m0,m0+32) x cols [0,64), K=64 ------------
    // preload all B fragments of the k tile once (reused by both M tiles)
    v16h bk[4][2];
    #pragma unroll
    for (int nt = 0; nt < 4; ++nt)
      #pragma unroll
      for (int ks = 0; ks < 2; ++ks)
        bk[nt][ks] = frag_b(sK + (size_t)(nt*16 + (lane & 15)) * PAD + ks*32, lane);

    #pragma unroll
    for (int mt = 0; mt < 2; ++mt) {
      v16h ai[2];
      #pragma unroll
      for (int ks = 0; ks < 2; ++ks)
        ai[ks] = frag_a(sI + (size_t)(m0 + mt*16 + (lane & 15)) * PAD + ks*32, lane);
      #pragma unroll
      for (int nt = 0; nt < 4; ++nt) {
        v8f c = wmma_f16(ai[0], bk[nt][0], vzero);
        c     = wmma_f16(ai[1], bk[nt][1], c);
        const int rbase = mt * 16 + ((lane & 16) ? 8 : 0);
        const int col   = nt * 16 + (lane & 15);
        #pragma unroll
        for (int e = 0; e < 8; ++e)
          sSw[(rbase + e) * PAD + col] = (_Float16)relu15f(c[e]);
      }
    }

    // --- Hm += S @ v : (M, D) 2x4 tiles, K=64 -----------------------------
    v16h bv[4][2];
    #pragma unroll
    for (int dt = 0; dt < 4; ++dt)
      #pragma unroll
      for (int ks = 0; ks < 2; ++ks)
        bv[dt][ks] = frag_b(sVT + (size_t)(dt*16 + (lane & 15)) * PAD + ks*32, lane);

    #pragma unroll
    for (int mt = 0; mt < 2; ++mt) {
      v16h as[2];
      #pragma unroll
      for (int ks = 0; ks < 2; ++ks)
        as[ks] = frag_a(sSw + (size_t)(mt*16 + (lane & 15)) * PAD + ks*32, lane);
      #pragma unroll
      for (int dt = 0; dt < 4; ++dt) {
        acc[mt][dt] = wmma_f16(as[0], bv[dt][0], acc[mt][dt]);
        acc[mt][dt] = wmma_f16(as[1], bv[dt][1], acc[mt][dt]);
      }
    }
  }

  // reduce partials across k-splits
  float* Hp = Hraw + (size_t)bh * LI_ * D_;
  #pragma unroll
  for (int mt = 0; mt < 2; ++mt) {
    #pragma unroll
    for (int dt = 0; dt < 4; ++dt) {
      const int rbase = m0 + mt * 16 + ((lane & 16) ? 8 : 0);
      const int col   = dt * 16 + (lane & 15);
      #pragma unroll
      for (int e = 0; e < 8; ++e)
        atomicAdd(Hp + (size_t)(rbase + e) * D_ + col, acc[mt][dt][e]);
    }
  }
}

// ---------------------------------------------------------------------------
// Linear: X[b,i,:] (= Hraw permuted) @ W^T + bias -> Hm2 (f16) and Hm2T (f16)
// X rows R = B*LI = 2048, cols 512. grid (4, 16), block 256. Dyn LDS 36864 B.
// ---------------------------------------------------------------------------
__global__ void __launch_bounds__(256) isab_linear(
    const float* __restrict__ Hraw, const float* __restrict__ W,
    const float* __restrict__ bias, _Float16* __restrict__ Hm2,
    _Float16* __restrict__ Hm2T)
{
  extern __shared__ _Float16 smem[];
  _Float16* sX = smem;              // [128][PAD]
  _Float16* sW = sX + 128 * PAD;    // [128][PAD]

  const int tid  = threadIdx.x;
  const int lane = tid & 31;
  const int wave = tid >> 5;
  const int wr   = wave >> 1;       // 0..3  (32-row groups)
  const int wc   = wave & 1;        // 0..1  (64-col groups)
  const int rbase = blockIdx.y * 128;
  const int cbase = blockIdx.x * 128;

  v8f acc[2][4];
  v8f vzero = {0.f,0.f,0.f,0.f,0.f,0.f,0.f,0.f};
  #pragma unroll
  for (int i = 0; i < 2; ++i)
    #pragma unroll
    for (int j = 0; j < 4; ++j) acc[i][j] = vzero;

  for (int kc = 0; kc < 8; ++kc) {          // K step = 64; h index == kc
    __syncthreads();
    for (int idx = tid; idx < 128 * 64; idx += 256) {
      int r = idx >> 6, c = idx & 63;
      int rg = rbase + r;
      int bb = rg >> 8, ii = rg & 255;
      sX[r * PAD + c] =
        (_Float16)Hraw[(((size_t)bb * 8 + kc) * 256 + ii) * 64 + c];
      sW[r * PAD + c] =
        (_Float16)W[(size_t)(cbase + r) * DH_ + kc * 64 + c];
    }
    __syncthreads();

    v16h bw[4][2];
    #pragma unroll
    for (int nt = 0; nt < 4; ++nt)
      #pragma unroll
      for (int ks = 0; ks < 2; ++ks)
        bw[nt][ks] = frag_b(sW + (size_t)(wc*64 + nt*16 + (lane & 15)) * PAD + ks*32, lane);

    #pragma unroll
    for (int mt = 0; mt < 2; ++mt) {
      v16h ax[2];
      #pragma unroll
      for (int ks = 0; ks < 2; ++ks)
        ax[ks] = frag_a(sX + (size_t)(wr*32 + mt*16 + (lane & 15)) * PAD + ks*32, lane);
      #pragma unroll
      for (int nt = 0; nt < 4; ++nt) {
        acc[mt][nt] = wmma_f16(ax[0], bw[nt][0], acc[mt][nt]);
        acc[mt][nt] = wmma_f16(ax[1], bw[nt][1], acc[mt][nt]);
      }
    }
  }

  // bias + scatter to Hm2 [bh][i][d] and Hm2T [bh][d][i]
  #pragma unroll
  for (int mt = 0; mt < 2; ++mt) {
    #pragma unroll
    for (int nt = 0; nt < 4; ++nt) {
      const int rloc = wr * 32 + mt * 16 + ((lane & 16) ? 8 : 0);
      const int o    = cbase + wc * 64 + nt * 16 + (lane & 15);
      const float bv = bias[o];
      const int hh = o >> 6, dd = o & 63;
      #pragma unroll
      for (int e = 0; e < 8; ++e) {
        int rr = rbase + rloc + e;
        int bb = rr >> 8, ii = rr & 255;
        float val = acc[mt][nt][e] + bv;
        size_t bhh = (size_t)bb * 8 + hh;
        Hm2 [(bhh * 256 + ii) * 64 + dd] = (_Float16)val;
        Hm2T[(bhh * 64 + dd) * 256 + ii] = (_Float16)val;
      }
    }
  }
}

// ---------------------------------------------------------------------------
// Stage 1: O = relu15((q/8) @ Hm2^T) @ Hm2
// grid (32, BH), block 256 (8 waves x 16 q-rows). Dyn LDS = 156672 B.
// ---------------------------------------------------------------------------
__global__ void __launch_bounds__(256) isab_stage1(
    const float* __restrict__ Q, const _Float16* __restrict__ Hm2,
    const _Float16* __restrict__ Hm2T, float* __restrict__ O)
{
  extern __shared__ _Float16 smem[];
  _Float16* sH  = smem;                // [256][PAD]   Hm2 rows (i, d)
  _Float16* sHT = sH  + 256 * PAD;     // [64][PADI]   Hm2T rows (d, i)
  _Float16* sQ  = sHT + 64 * PADI;     // [128][PAD]   q/8 tile
  _Float16* sS  = sQ  + 128 * PAD;     // [8][16][PADI] per-wave attn rows

  const int bh   = blockIdx.y;
  const int q0   = blockIdx.x * 128;
  const int tid  = threadIdx.x;
  const int lane = tid & 31;
  const int wave = tid >> 5;

  // stage induced set, both layouts (already f16 in global)
  const _Float16* Hp = Hm2 + (size_t)bh * 256 * 64;
  for (int idx = tid; idx < (256 * 64) / 8; idx += 256) {
    int r = idx >> 3, c8 = (idx & 7) * 8;
    *(h8*)(sH + r * PAD + c8) = *(const h8*)(Hp + r * 64 + c8);
  }
  const _Float16* HTp = Hm2T + (size_t)bh * 64 * 256;
  for (int idx = tid; idx < (64 * 256) / 8; idx += 256) {
    int r = idx >> 5, c8 = (idx & 31) * 8;
    *(h8*)(sHT + r * PADI + c8) = *(const h8*)(HTp + r * 256 + c8);
  }
  // stage q/8 tile
  const float* Qp = Q + ((size_t)bh * LQ_ + q0) * D_;
  for (int idx = tid; idx < 128 * 64; idx += 256) {
    int r = idx >> 6, c = idx & 63;
    sQ[r * PAD + c] = (_Float16)(Qp[idx] * 0.125f);
  }
  __syncthreads();

  v8f vzero = {0.f,0.f,0.f,0.f,0.f,0.f,0.f,0.f};
  _Float16* sSw = sS + (size_t)wave * 16 * PADI;

  // S = (q/8) @ Hm2^T : 16 rows x 256 cols, K = 64
  // A fragments are invariant across all 16 N tiles -> hoist.
  v16h aq[2];
  #pragma unroll
  for (int ks = 0; ks < 2; ++ks)
    aq[ks] = frag_a(sQ + (size_t)(wave*16 + (lane & 15)) * PAD + ks*32, lane);

  #pragma unroll
  for (int nt = 0; nt < 16; ++nt) {
    v16h b0 = frag_b(sH + (size_t)(nt*16 + (lane & 15)) * PAD, lane);
    v16h b1 = frag_b(sH + (size_t)(nt*16 + (lane & 15)) * PAD + 32, lane);
    v8f c = wmma_f16(aq[0], b0, vzero);
    c     = wmma_f16(aq[1], b1, c);
    const int rbase = (lane & 16) ? 8 : 0;
    const int col   = nt * 16 + (lane & 15);
    #pragma unroll
    for (int e = 0; e < 8; ++e)
      sSw[(rbase + e) * PADI + col] = (_Float16)relu15f(c[e]);
  }

  // O = S @ Hm2 : 16 rows x 64 cols, K = 256
  v8f acc[4] = {vzero, vzero, vzero, vzero};
  #pragma unroll
  for (int ks = 0; ks < 8; ++ks) {
    v16h a = frag_a(sSw + (size_t)(lane & 15) * PADI + ks*32, lane);
    #pragma unroll
    for (int dt = 0; dt < 4; ++dt) {
      v16h b = frag_b(sHT + (size_t)(dt*16 + (lane & 15)) * PADI + ks*32, lane);
      acc[dt] = wmma_f16(a, b, acc[dt]);
    }
  }

  float* Op = O + ((size_t)bh * LQ_ + q0 + wave * 16) * D_;
  #pragma unroll
  for (int dt = 0; dt < 4; ++dt) {
    const int rbase = (lane & 16) ? 8 : 0;
    const int col   = dt * 16 + (lane & 15);
    #pragma unroll
    for (int e = 0; e < 8; ++e)
      Op[(size_t)(rbase + e) * D_ + col] = acc[dt][e];
  }
}

// ---------------------------------------------------------------------------
// Host launcher
// ---------------------------------------------------------------------------
extern "C" void kernel_launch(void* const* d_in, const int* in_sizes, int n_in,
                              void* d_out, int out_size, void* d_ws, size_t ws_size,
                              hipStream_t stream) {
  (void)in_sizes; (void)n_in; (void)out_size; (void)ws_size;
  const float* q  = (const float*)d_in[0];
  const float* k  = (const float*)d_in[1];
  const float* v  = (const float*)d_in[2];
  const float* I  = (const float*)d_in[3];
  const float* w  = (const float*)d_in[4];
  const float* wb = (const float*)d_in[5];
  float* out = (float*)d_out;

  char* ws = (char*)d_ws;
  float*    Hraw = (float*)ws;                                 // 4 MiB f32
  _Float16* Hm2  = (_Float16*)(ws + (size_t)4 * 1024 * 1024);  // 2 MiB f16
  _Float16* Hm2T = (_Float16*)(ws + (size_t)6 * 1024 * 1024);  // 2 MiB f16

  const int lds0 = (LI_*PAD + KT*PAD + KT*PAD + 8*32*PAD) * 2;      // 92160
  const int lds1 = (128*PAD + 128*PAD) * 2;                          // 36864
  const int lds2 = (256*PAD + 64*PADI + 128*PAD + 8*16*PADI) * 2;    // 156672

  // allow >64KB dynamic LDS (idempotent; called every launch)
  (void)hipFuncSetAttribute((const void*)isab_stage0,
      hipFuncAttributeMaxDynamicSharedMemorySize, lds0);
  (void)hipFuncSetAttribute((const void*)isab_linear,
      hipFuncAttributeMaxDynamicSharedMemorySize, lds1);
  (void)hipFuncSetAttribute((const void*)isab_stage1,
      hipFuncAttributeMaxDynamicSharedMemorySize, lds2);

  // zero partial buffer (atomically accumulated in stage 0)
  (void)hipMemsetAsync(Hraw, 0, (size_t)BH_ * LI_ * D_ * sizeof(float), stream);

  isab_stage0<<<dim3(KSPLIT, BH_), 256, lds0, stream>>>(I, k, v, Hraw);
  isab_linear<<<dim3(4, 16),       256, lds1, stream>>>(Hraw, w, wb, Hm2, Hm2T);
  isab_stage1<<<dim3(32, BH_),     256, lds2, stream>>>(q, Hm2, Hm2T, out);
}
